// BidirectionalAttention_20710332301677
// MI455X (gfx1250) — matrix operand order repaired and checked by reading, not compiled, not run on hardware
//
#include <hip/hip_runtime.h>
#include <hip/hip_bf16.h>

typedef __bf16 bf16;
typedef __attribute__((ext_vector_type(8)))  bf16  v8bf;
typedef __attribute__((ext_vector_type(16))) bf16  v16bf;
typedef __attribute__((ext_vector_type(8)))  float v8f;

constexpr int BATCH = 4;
constexpr int TSEQ  = 2048;
constexpr int CDIM  = 1024;
constexpr int NH    = 16;
constexpr int DH    = 64;
constexpr int MROWS = BATCH * TSEQ;   // 8192
constexpr int QB    = 2;              // q-tiles per wave in attention

static __device__ __forceinline__ v16bf cat16(v8bf a, v8bf b) {
  return __builtin_shufflevector(a, b, 0,1,2,3,4,5,6,7,8,9,10,11,12,13,14,15);
}
static __device__ __forceinline__ v8bf ld8(const bf16* p) { return *(const v8bf*)p; }

static __device__ __forceinline__ v8f wmma_bf16(v16bf a, v16bf b, v8f c) {
  // D = A(16x32 bf16) * B(32x16 bf16) + C(16x16 f32)
  return __builtin_amdgcn_wmma_f32_16x16x32_bf16(false, a, false, b, (short)0, c, false, false);
}

// ---------------------------------------------------------------- convert f32 -> bf16
__global__ void cvt_f32_bf16(const float* __restrict__ in, bf16* __restrict__ out, int n) {
  int i = blockIdx.x * blockDim.x + threadIdx.x;
  int stride = gridDim.x * blockDim.x;
  for (; i < n; i += stride) out[i] = (bf16)in[i];
}

// ---------------------------------------------------------------- QKV GEMM (register-blocked 32x64 per wave)
// Xb [M=8192, K=1024] (K-contig) x Wb [N=3072, K=1024] (K-contig) -> scatter Q/K/Vt (bf16)
__global__ __launch_bounds__(256) void qkv_gemm(const bf16* __restrict__ Xb,
                                                const bf16* __restrict__ Wb,
                                                bf16* __restrict__ Qb,
                                                bf16* __restrict__ Kb,
                                                bf16* __restrict__ Vt) {
  const int lane = threadIdx.x & 31;
  const int wave = threadIdx.x >> 5;
  const int gw   = blockIdx.x * 8 + wave;           // global wave id
  const int NTN  = (3 * CDIM) / 64;                 // 48 N-tiles (64 wide)
  const int tm   = gw / NTN;                        // 0..255 (32-row tiles)
  const int tn   = gw % NTN;
  const int r    = lane & 15;
  const int hf   = lane >> 4;

  const bf16* Arow0 = Xb + (size_t)(tm * 32 + r) * CDIM;
  const bf16* Arow1 = Arow0 + (size_t)16 * CDIM;
  const bf16* Brow[4];
#pragma unroll
  for (int j = 0; j < 4; ++j)
    Brow[j] = Wb + (size_t)(tn * 64 + 16 * j + r) * CDIM;

  v8f acc[2][4];
#pragma unroll
  for (int mi = 0; mi < 2; ++mi)
#pragma unroll
    for (int j = 0; j < 4; ++j) acc[mi][j] = (v8f){0.f,0.f,0.f,0.f,0.f,0.f,0.f,0.f};

  for (int k0 = 0; k0 < CDIM; k0 += 32) {
    if (k0 + 256 < CDIM) {
      __builtin_prefetch(Arow0 + k0 + 256, 0, 1);
      __builtin_prefetch(Arow1 + k0 + 256, 0, 1);
    }
    v16bf a0 = cat16(ld8(Arow0 + k0 + hf * 8), ld8(Arow0 + k0 + 16 + hf * 8));
    v16bf a1 = cat16(ld8(Arow1 + k0 + hf * 8), ld8(Arow1 + k0 + 16 + hf * 8));
#pragma unroll
    for (int j = 0; j < 4; ++j) {
      v16bf bfr = cat16(ld8(Brow[j] + k0 + hf * 8), ld8(Brow[j] + k0 + 16 + hf * 8));
      acc[0][j] = wmma_bf16(a0, bfr, acc[0][j]);
      acc[1][j] = wmma_bf16(a1, bfr, acc[1][j]);
    }
  }

  // Epilogue scatter. 8 sub-tiles of 16x16.
#pragma unroll
  for (int mi = 0; mi < 2; ++mi) {
#pragma unroll
    for (int j = 0; j < 4; ++j) {
      const int n      = tn * 64 + 16 * j + r;
      const int part   = n / CDIM;
      const int within = n % CDIM;
      const int h      = within / DH;
      const int d      = within % DH;
#pragma unroll
      for (int i = 0; i < 8; ++i) {
        const int m = tm * 32 + mi * 16 + hf * 8 + i;
        const int b = m / TSEQ, t = m % TSEQ;
        const bf16 v = (bf16)acc[mi][j][i];
        if (part == 0)      Qb[(((size_t)(b * NH + h)) * TSEQ + t) * DH + d] = v;
        else if (part == 1) Kb[(((size_t)(b * NH + h)) * TSEQ + t) * DH + d] = v;
        else                Vt[(((size_t)(b * NH + h)) * DH + d) * TSEQ + t] = v;  // transposed
      }
    }
  }
}

// ---------------------------------------------------------------- flash attention
// One wave per 2 q-tiles (32 queries). Keys streamed in chunks of 32;
// K/V fragments are shared across both q-tiles (2x arithmetic intensity).
__global__ __launch_bounds__(128) void attn_kernel(const bf16* __restrict__ Qb,
                                                   const bf16* __restrict__ Kb,
                                                   const bf16* __restrict__ Vt,
                                                   const unsigned char* __restrict__ mask,
                                                   bf16* __restrict__ AO) {
  __shared__ float Sbuf[4][QB][16][32];
  __shared__ bf16  Pbuf[4][QB][16][32];
  __shared__ float rowmax[4][QB][16];
  __shared__ float rowsum[4][QB][16];
  __shared__ float rowcorr[4][QB][16];

  const int lane = threadIdx.x & 31;
  const int wave = threadIdx.x >> 5;
  const int gw   = blockIdx.x * 4 + wave;
  const int NQG  = TSEQ / (16 * QB);     // 64 q-groups
  const int qg   = gw % NQG;
  const int bh   = gw / NQG;             // b*NH + h
  const int b    = bh / NH;
  const int h    = bh % NH;
  const int r    = lane & 15;
  const int hf   = lane >> 4;
  const float scale = 0.125f;            // 1/sqrt(64)

  // Q A-fragments for each q-tile (contraction over d, two 32-wide chunks)
  v16bf qfrag[QB][2];
#pragma unroll
  for (int q = 0; q < QB; ++q) {
    const bf16* Qrow = Qb + ((size_t)bh * TSEQ + (qg * QB + q) * 16 + r) * DH;
#pragma unroll
    for (int dc = 0; dc < 2; ++dc)
      qfrag[q][dc] = cat16(ld8(Qrow + dc * 32 + hf * 8), ld8(Qrow + dc * 32 + 16 + hf * 8));
  }

  v8f oacc[QB][4];
#pragma unroll
  for (int q = 0; q < QB; ++q)
#pragma unroll
    for (int dt = 0; dt < 4; ++dt) oacc[q][dt] = (v8f){0.f,0.f,0.f,0.f,0.f,0.f,0.f,0.f};

  if (lane < 16) {
#pragma unroll
    for (int q = 0; q < QB; ++q) {
      rowmax[wave][q][lane] = -__builtin_inff();
      rowsum[wave][q][lane] = 0.f;
    }
  }
  __syncthreads();

  const unsigned char* mrow = mask + (size_t)b * TSEQ;

  for (int kc = 0; kc < TSEQ; kc += 32) {
    // ---- S = Q K^T for 32 keys, both q-tiles, sharing K fragments
#pragma unroll
    for (int sub = 0; sub < 2; ++sub) {
      v8f s[QB];
#pragma unroll
      for (int q = 0; q < QB; ++q) s[q] = (v8f){0.f,0.f,0.f,0.f,0.f,0.f,0.f,0.f};
      const bf16* Krow = Kb + ((size_t)bh * TSEQ + kc + sub * 16 + r) * DH;  // col = key
#pragma unroll
      for (int dc = 0; dc < 2; ++dc) {
        v16bf kf = cat16(ld8(Krow + dc * 32 + hf * 8), ld8(Krow + dc * 32 + 16 + hf * 8));
#pragma unroll
        for (int q = 0; q < QB; ++q) s[q] = wmma_bf16(qfrag[q][dc], kf, s[q]);
      }
#pragma unroll
      for (int q = 0; q < QB; ++q)
#pragma unroll
        for (int i = 0; i < 8; ++i) Sbuf[wave][q][hf * 8 + i][sub * 16 + r] = s[q][i];
    }
    __syncthreads();

    // ---- online softmax (mask bias shared across q-tiles)
    float mbias[16];
#pragma unroll
    for (int c = 0; c < 16; ++c)
      mbias[c] = mrow[kc + hf * 16 + c] ? -__builtin_inff() : 0.f;

#pragma unroll
    for (int q = 0; q < QB; ++q) {
      float vals[16];
      float cmax = -__builtin_inff();
#pragma unroll
      for (int c = 0; c < 16; ++c) {
        const float v = Sbuf[wave][q][r][hf * 16 + c] * scale + mbias[c];
        vals[c] = v;
        cmax = fmaxf(cmax, v);
      }
      cmax = fmaxf(cmax, __shfl_xor(cmax, 16, 32));
      const float oldmax = rowmax[wave][q][r];
      const float newmax = fmaxf(oldmax, cmax);
      const bool  dead   = (newmax == -__builtin_inff());
      const float corr   = dead ? 1.0f : __expf(oldmax - newmax);
      float csum = 0.f;
#pragma unroll
      for (int c = 0; c < 16; ++c) {
        const float p = dead ? 0.f : __expf(vals[c] - newmax);
        csum += p;
        Pbuf[wave][q][r][hf * 16 + c] = (bf16)p;
      }
      csum += __shfl_xor(csum, 16, 32);
      if (hf == 0) {
        rowmax[wave][q][r]  = newmax;
        rowsum[wave][q][r]  = rowsum[wave][q][r] * corr + csum;
        rowcorr[wave][q][r] = corr;
      }
    }
    __syncthreads();

    // ---- O = O*corr + P V, sharing V fragments across q-tiles
    v16bf pfrag[QB];
    float cr[QB][8];
#pragma unroll
    for (int q = 0; q < QB; ++q) {
      const bf16* Prow = &Pbuf[wave][q][r][0];
      pfrag[q] = cat16(ld8(Prow + hf * 8), ld8(Prow + 16 + hf * 8));
#pragma unroll
      for (int i = 0; i < 8; ++i) cr[q][i] = rowcorr[wave][q][hf * 8 + i];
    }
#pragma unroll
    for (int dt = 0; dt < 4; ++dt) {
      const bf16* Vrow = Vt + ((size_t)bh * DH + dt * 16 + r) * TSEQ + kc;  // col = d
      v16bf vf = cat16(ld8(Vrow + hf * 8), ld8(Vrow + 16 + hf * 8));
#pragma unroll
      for (int q = 0; q < QB; ++q) {
#pragma unroll
        for (int i = 0; i < 8; ++i) oacc[q][dt][i] *= cr[q][i];
        oacc[q][dt] = wmma_bf16(pfrag[q], vf, oacc[q][dt]);
      }
    }
  }
  __syncthreads();

  // ---- epilogue: normalize and store [b, t, h*64 + d] as bf16
#pragma unroll
  for (int q = 0; q < QB; ++q) {
    float inv[8];
#pragma unroll
    for (int i = 0; i < 8; ++i) {
      const float s = rowsum[wave][q][hf * 8 + i];
      inv[i] = (s > 0.f) ? (1.f / s) : 0.f;
    }
#pragma unroll
    for (int dt = 0; dt < 4; ++dt)
#pragma unroll
      for (int i = 0; i < 8; ++i) {
        const int t = (qg * QB + q) * 16 + hf * 8 + i;
        AO[((size_t)b * TSEQ + t) * CDIM + h * DH + dt * 16 + r] = (bf16)(oacc[q][dt][i] * inv[i]);
      }
  }
}

// ---------------------------------------------------------------- output projection (register-blocked 32x64 per wave)
// AO [M=8192, K=1024] x Wproj [N=1024, K=1024] -> f32 out [M, N]
__global__ __launch_bounds__(256) void proj_gemm(const bf16* __restrict__ Ab,
                                                 const bf16* __restrict__ Wb,
                                                 float* __restrict__ out) {
  const int lane = threadIdx.x & 31;
  const int wave = threadIdx.x >> 5;
  const int gw   = blockIdx.x * 8 + wave;
  const int NTN  = CDIM / 64;                 // 16 N-tiles (64 wide)
  const int tm   = gw / NTN;
  const int tn   = gw % NTN;
  const int r    = lane & 15;
  const int hf   = lane >> 4;

  const bf16* Arow0 = Ab + (size_t)(tm * 32 + r) * CDIM;
  const bf16* Arow1 = Arow0 + (size_t)16 * CDIM;
  const bf16* Brow[4];
#pragma unroll
  for (int j = 0; j < 4; ++j)
    Brow[j] = Wb + (size_t)(tn * 64 + 16 * j + r) * CDIM;

  v8f acc[2][4];
#pragma unroll
  for (int mi = 0; mi < 2; ++mi)
#pragma unroll
    for (int j = 0; j < 4; ++j) acc[mi][j] = (v8f){0.f,0.f,0.f,0.f,0.f,0.f,0.f,0.f};

  for (int k0 = 0; k0 < CDIM; k0 += 32) {
    if (k0 + 256 < CDIM) {
      __builtin_prefetch(Arow0 + k0 + 256, 0, 1);
      __builtin_prefetch(Arow1 + k0 + 256, 0, 1);
    }
    v16bf a0 = cat16(ld8(Arow0 + k0 + hf * 8), ld8(Arow0 + k0 + 16 + hf * 8));
    v16bf a1 = cat16(ld8(Arow1 + k0 + hf * 8), ld8(Arow1 + k0 + 16 + hf * 8));
#pragma unroll
    for (int j = 0; j < 4; ++j) {
      v16bf bfr = cat16(ld8(Brow[j] + k0 + hf * 8), ld8(Brow[j] + k0 + 16 + hf * 8));
      acc[0][j] = wmma_bf16(a0, bfr, acc[0][j]);
      acc[1][j] = wmma_bf16(a1, bfr, acc[1][j]);
    }
  }
#pragma unroll
  for (int mi = 0; mi < 2; ++mi)
#pragma unroll
    for (int j = 0; j < 4; ++j)
#pragma unroll
      for (int i = 0; i < 8; ++i) {
        const int m = tm * 32 + mi * 16 + hf * 8 + i;
        out[(size_t)m * CDIM + tn * 64 + 16 * j + r] = acc[mi][j][i];
      }
}

// ---------------------------------------------------------------- launch
extern "C" void kernel_launch(void* const* d_in, const int* in_sizes, int n_in,
                              void* d_out, int out_size, void* d_ws, size_t ws_size,
                              hipStream_t stream) {
  const float*         x     = (const float*)d_in[0];
  const unsigned char* mask  = (const unsigned char*)d_in[1];  // bool key-padding mask
  const float*         Wqkv  = (const float*)d_in[2];
  const float*         Wproj = (const float*)d_in[3];
  float*               out   = (float*)d_out;

  char* ws = (char*)d_ws;
  const size_t MiB = 1u << 20;
  bf16* xb     = (bf16*)(ws + 0 * MiB);    // 8192*1024   bf16 = 16 MiB
  bf16* wqkvb  = (bf16*)(ws + 16 * MiB);   // 3072*1024   bf16 =  6 MiB
  bf16* wprojb = (bf16*)(ws + 22 * MiB);   // 1024*1024   bf16 =  2 MiB
  bf16* Qb     = (bf16*)(ws + 24 * MiB);   // [b,h,t,d]         16 MiB
  bf16* Kb     = (bf16*)(ws + 40 * MiB);   // [b,h,t,d]         16 MiB
  bf16* Vt     = (bf16*)(ws + 56 * MiB);   // [b,h,d,t]         16 MiB
  bf16* AO     = (bf16*)(ws + 72 * MiB);   // [b,t,h*d]         16 MiB (ends at 88 MiB)

  cvt_f32_bf16<<<4096, 256, 0, stream>>>(x,     xb,     MROWS * CDIM);
  cvt_f32_bf16<<<2048, 256, 0, stream>>>(Wqkv,  wqkvb,  3 * CDIM * CDIM);
  cvt_f32_bf16<<<1024, 256, 0, stream>>>(Wproj, wprojb, CDIM * CDIM);

  // (8192/32) * (3072/64) = 12288 waves / 8 per block
  qkv_gemm<<<1536, 256, 0, stream>>>(xb, wqkvb, Qb, Kb, Vt);

  // B*H*(T/32) = 4096 waves / 4 per block
  attn_kernel<<<1024, 128, 0, stream>>>(Qb, Kb, Vt, mask, AO);

  // (8192/32) * (1024/64) = 4096 waves / 8 per block
  proj_gemm<<<512, 256, 0, stream>>>(AO, wprojb, out);
}
